// BasicRNNBlock_979252544170
// MI455X (gfx1250) — compile-verified, matrix operand order checked by reading, not run on hardware
//
#include <hip/hip_runtime.h>

// Vanilla tanh RNN: B=64, T=512, I=1024, H=1024.
// Phase 0: convert W_ih / W_hh to bf16 (one-time, weights L2-resident after).
// Phase 1: xp = x * W_ih^T + (b_ih + b_hh), bf16 WMMA GEMM, written into d_out[b,t,h].
// Phase 2: persistent recurrence kernel, 64 WGs (4 batch groups x 16 column slices),
//          h tile staged LDS via TDM tensor_load_to_lds (TENSORcnt), per-step 16x64
//          tile of h*W_hh^T via v_wmma_f32_16x16x32_bf16, tanh epilogue, inter-WG
//          sync via monotonic atomic counter (per batch group) + s_sleep spin.

#define B_ 64
#define T_ 512
#define I_ 1024
#define H_ 1024

typedef __attribute__((ext_vector_type(16))) __bf16 v16bf;
typedef __attribute__((ext_vector_type(8)))  float  v8f;
typedef unsigned int u32x4 __attribute__((ext_vector_type(4)));
typedef unsigned int u32x8 __attribute__((ext_vector_type(8)));

__device__ __forceinline__ unsigned short f2bf_raw(float f) {
  union { float f; unsigned u; } x; x.f = f;
  unsigned r = x.u + 0x7FFFu + ((x.u >> 16) & 1u);  // round-to-nearest-even
  return (unsigned short)(r >> 16);
}
__device__ __forceinline__ unsigned pack2(float lo, float hi) {
  return (unsigned)f2bf_raw(lo) | ((unsigned)f2bf_raw(hi) << 16);
}

// 16-bit A/B fragment (ISA 7.12.2): lane half = lane>>4.
// elements 0..7  = K = kbase + 8*half + j
// elements 8..15 = K = kbase + 16 + 8*half + j
// -> two contiguous 16B chunks at (p) and (p + 16 elems).
__device__ __forceinline__ v16bf load_frag(const unsigned short* p) {
  union { v16bf v; uint4 q[2]; } f;
  f.q[0] = *(const uint4*)(p);
  f.q[1] = *(const uint4*)(p + 16);
  return f.v;
}

// ---------------- Phase 0: weight conversion f32 -> bf16 ----------------
__global__ void __launch_bounds__(256) rnn_cvt_kernel(
    const float* __restrict__ wih, const float* __restrict__ whh,
    unsigned short* __restrict__ wihb, unsigned short* __restrict__ whhb) {
  const int i = blockIdx.x * 256 + threadIdx.x;   // grid covers H*I == H*H elements
  wihb[i] = f2bf_raw(wih[i]);
  whhb[i] = f2bf_raw(whh[i]);
}

// zero h0 double-buffer and the 4 group barrier counters (every launch -> deterministic)
__global__ void __launch_bounds__(256) rnn_init_kernel(
    unsigned short* __restrict__ hbuf, unsigned* __restrict__ cnt) {
  const int i = blockIdx.x * 256 + threadIdx.x;
  if (i < 2 * B_ * H_) hbuf[i] = 0;
  if (i < 4) cnt[i] = 0;
}

// ---------------- Phase 1: xp projection GEMM ----------------
// Grid: (M/64, H/32), 256 threads = 8 waves; macro tile 64x32, wave tile 16x16.
__global__ void __launch_bounds__(256) rnn_xproj_kernel(
    const float* __restrict__ x, const unsigned short* __restrict__ wih,
    const float* __restrict__ b_ih, const float* __restrict__ b_hh,
    float* __restrict__ out) {
  __shared__ unsigned short Ash[64][72];   // +8 pad: spread DS banks, keep 16B align
  __shared__ unsigned short Bsh[32][72];   // stored [n][k], k contiguous
  const int tid = threadIdx.x;
  const int rowBase = blockIdx.x * 64;     // row r = b*T + t (x and out share row index)
  const int colBase = blockIdx.y * 32;
  const int wave = tid >> 5, lane = tid & 31;
  const int half = lane >> 4, ln = lane & 15;
  const int mt = wave >> 1, nt = wave & 1;

  const int hcol = colBase + nt * 16 + ln;
  const float bias = b_ih[hcol] + b_hh[hcol];

  v8f acc = {};

  const int ar = tid >> 2, ak = (tid & 3) * 16;   // A stage: 64 rows x 64 k f32
  const int bn = tid >> 3, bk = (tid & 7) * 8;    // B stage: 32 rows x 64 k bf16
  const float* xrow = x + (size_t)(rowBase + ar) * I_;
  const unsigned short* wrow = wih + (size_t)(colBase + bn) * I_;

  for (int k0 = 0; k0 < I_; k0 += 64) {
    const float* s = xrow + k0 + ak;
    float4 f0 = *(const float4*)(s);
    float4 f1 = *(const float4*)(s + 4);
    float4 f2 = *(const float4*)(s + 8);
    float4 f3 = *(const float4*)(s + 12);
    uint4 q0, q1;
    q0.x = pack2(f0.x, f0.y); q0.y = pack2(f0.z, f0.w);
    q0.z = pack2(f1.x, f1.y); q0.w = pack2(f1.z, f1.w);
    q1.x = pack2(f2.x, f2.y); q1.y = pack2(f2.z, f2.w);
    q1.z = pack2(f3.x, f3.y); q1.w = pack2(f3.z, f3.w);
    *(uint4*)&Ash[ar][ak]     = q0;
    *(uint4*)&Ash[ar][ak + 8] = q1;
    *(uint4*)&Bsh[bn][bk] = *(const uint4*)(wrow + k0 + bk);
    __syncthreads();
#pragma unroll
    for (int kk = 0; kk < 64; kk += 32) {
      v16bf a = load_frag(&Ash[mt * 16 + ln][kk + 8 * half]);
      v16bf b = load_frag(&Bsh[nt * 16 + ln][kk + 8 * half]);
      acc = __builtin_amdgcn_wmma_f32_16x16x32_bf16(false, a, false, b,
                                                    (short)0, acc, false, false);
    }
    __syncthreads();
  }
  // C layout: VGPR v -> M = v + 8*half, N = ln
  const int r0 = rowBase + mt * 16 + 8 * half;
#pragma unroll
  for (int v = 0; v < 8; ++v) {
    out[(size_t)(r0 + v) * H_ + hcol] = acc[v] + bias;
  }
}

// ---------------- Phase 2: persistent recurrence ----------------
// 64 WGs x 128 threads. WG (g,s): batch rows g*16..g*16+15, columns s*64..s*64+63.
// 4 waves, one 16x16 n-tile each; K=1024 loop = 32 wmma per wave per timestep.
// h tile (16x1024 bf16, 32KB) staged into LDS by the Tensor Data Mover each step.
// TDM pad: pad_interval=256 DWORDs (val 7), pad_amount=4 DWORDs (val 3)
//   -> LDS byte addr of element (m,k) = m*2080 + 2*k + 16*(k>>9); rows spread banks.
#define HROW_ 1040   // padded row stride in elements (2080 bytes)

__global__ void __launch_bounds__(128) rnn_recur_kernel(
    const unsigned short* __restrict__ whh,
    unsigned short* __restrict__ hbuf,    // double-buffered [2][B][H] bf16
    unsigned* __restrict__ cnt,           // 4 monotonic barrier counters
    float* __restrict__ out) {            // holds xp on input, h on output
  __shared__ unsigned short Hsh[16 * HROW_ + 16];
  const int tid = threadIdx.x;
  const int g = blockIdx.x >> 4;
  const int s = blockIdx.x & 15;
  const int wave = tid >> 5, lane = tid & 31;
  const int half = lane >> 4, ln = lane & 15;
  const int hcol = s * 64 + wave * 16 + ln;
  const unsigned short* wrow = whh + (size_t)hcol * H_;  // B[k][n] = W_hh[hcol][k]

  // ---- TDM descriptor (D#) constant parts: 2-D tile 1024 x 16, data_size=2B ----
  const unsigned lds_base = (unsigned)(size_t)(void*)&Hsh[0];
  u32x8 g1;
  g1[0] = (1u << 16)          // data_size = 1 -> 2 bytes
        | (1u << 20)          // pad_enable
        | (7u << 22)          // pad_interval: 256 DWORDs
        | (3u << 25);         // pad_amount: 4 DWORDs (16B)
  g1[1] = (1024u & 0xFFFFu) << 16;     // tensor_dim0[15:0] = 1024
  g1[2] = (16u << 16);                  // tensor_dim0 hi = 0, tensor_dim1 = 16
  g1[3] = (1024u << 16);                // tensor_dim1 hi = 0, tile_dim0 = 1024
  g1[4] = 16u;                          // tile_dim1 = 16, tile_dim2 = 0 (2-D)
  g1[5] = 1024u;                        // tensor_dim0_stride lo = 1024
  g1[6] = 0u;                           // stride hi, tensor_dim1_stride lo
  g1[7] = 0u;
  const u32x4 gz = {0u, 0u, 0u, 0u};    // groups 2/3 unused (tile_dim2/3 = 0)

  for (int t = 0; t < T_; ++t) {
    const unsigned short* hcur = hbuf + (size_t)(t & 1) * (B_ * H_);
    unsigned short*       hnxt = hbuf + (size_t)((t + 1) & 1) * (B_ * H_);

    // ---- stage this batch group's h rows via Tensor Data Mover ----
    if ((tid >> 5) == 0) {
      const unsigned long long ga =
          (unsigned long long)(size_t)(hcur + (size_t)(g * 16) * H_);
      u32x4 g0;
      g0[0] = 1u;                                   // count = 1 (valid descriptor)
      g0[1] = lds_base;                             // lds_addr
      g0[2] = (unsigned)ga;                         // global_addr[31:0]
      g0[3] = ((unsigned)(ga >> 32) & 0x01FFFFFFu)  // global_addr[56:32]
            | (2u << 30);                           // type = 2 ("image")
      asm volatile("tensor_load_to_lds %0, %1, %2, %3"
                   :: "s"(g0), "s"(g1), "s"(gz), "s"(gz) : "memory");
      __builtin_amdgcn_s_wait_tensorcnt(0);
    }
    __syncthreads();

    v8f acc = {};
#pragma unroll 4
    for (int k0 = 0; k0 < H_; k0 += 32) {
      const int ka = k0 + 8 * half;
      v16bf a = load_frag(&Hsh[ln * HROW_ + ka + 8 * (ka >> 9)]);  // ds_load_b128 x2
      v16bf b = load_frag(wrow + k0 + 8 * half);   // global_load_b128 x2 (L2-resident)
      if (k0 + 64 < H_) __builtin_prefetch((const void*)(wrow + k0 + 64), 0, 0);
      acc = __builtin_amdgcn_wmma_f32_16x16x32_bf16(false, a, false, b,
                                                    (short)0, acc, false, false);
    }

    // epilogue: h_new = tanh(acc + xp); overwrite xp slot in d_out; publish bf16 h
#pragma unroll
    for (int v = 0; v < 8; ++v) {
      const int b = g * 16 + v + 8 * half;
      const size_t idx = ((size_t)b * T_ + t) * H_ + hcol;
      const float hn = tanhf(acc[v] + out[idx]);
      out[idx] = hn;
      hnxt[(size_t)b * H_ + hcol] = f2bf_raw(hn);
    }

    // release published h, then barrier across the 16 WGs of this batch group
    __builtin_amdgcn_fence(__ATOMIC_RELEASE, "agent");
    __syncthreads();
    if (tid == 0) {
      atomicAdd(&cnt[g], 1u);
      const unsigned target = 16u * (unsigned)(t + 1);
      while (__atomic_load_n(&cnt[g], __ATOMIC_RELAXED) < target) {
        __builtin_amdgcn_s_sleep(2);
      }
    }
    __syncthreads();
    __builtin_amdgcn_fence(__ATOMIC_ACQUIRE, "agent");
  }
}

extern "C" void kernel_launch(void* const* d_in, const int* in_sizes, int n_in,
                              void* d_out, int out_size, void* d_ws, size_t ws_size,
                              hipStream_t stream) {
  (void)in_sizes; (void)n_in; (void)out_size; (void)ws_size;
  const float* x    = (const float*)d_in[0];
  const float* W_ih = (const float*)d_in[1];
  const float* W_hh = (const float*)d_in[2];
  const float* b_ih = (const float*)d_in[3];
  const float* b_hh = (const float*)d_in[4];
  float* out = (float*)d_out;

  char* ws = (char*)d_ws;
  unsigned short* wihb = (unsigned short*)ws;                                  // 2 MB
  unsigned short* whhb = (unsigned short*)(ws + (size_t)H_ * I_ * 2);          // 2 MB
  unsigned short* hbuf = (unsigned short*)(ws + (size_t)H_ * I_ * 2
                                              + (size_t)H_ * H_ * 2);          // 256 KB
  unsigned* cnt = (unsigned*)(ws + (size_t)H_ * I_ * 2 + (size_t)H_ * H_ * 2
                                 + (size_t)2 * B_ * H_ * 2);                   // 16 B

  rnn_cvt_kernel<<<(H_ * I_) / 256, 256, 0, stream>>>(W_ih, W_hh, wihb, whhb);
  rnn_init_kernel<<<(2 * B_ * H_) / 256, 256, 0, stream>>>(hbuf, cnt);
  rnn_xproj_kernel<<<dim3((B_ * T_) / 64, H_ / 32), 256, 0, stream>>>(
      x, wihb, b_ih, b_hh, out);
  rnn_recur_kernel<<<64, 128, 0, stream>>>(whhb, hbuf, cnt, out);
}
